// Architecture_46531675685184
// MI455X (gfx1250) — matrix operand order repaired and checked by reading, not compiled, not run on hardware
//
#include <hip/hip_runtime.h>
#include <hip/hip_bf16.h>

#define BB   64
#define SS   512
#define DDIM 256
#define HH   8
#define DFFN 1024
#define LLAY 6
#define NROWS (BB * SS)   // 32768
#define DKH  32

typedef __bf16 v16bf __attribute__((ext_vector_type(16)));
typedef __bf16 v8bf  __attribute__((ext_vector_type(8)));
typedef float  v8f   __attribute__((ext_vector_type(8)));

union ABfrag { v16bf v; v8bf h[2]; };

__device__ __forceinline__ float redmax16(float v) {
#pragma unroll
  for (int m = 1; m < 16; m <<= 1) v = fmaxf(v, __shfl_xor(v, m, 32));
  return v;
}
__device__ __forceinline__ float redsum16(float v) {
#pragma unroll
  for (int m = 1; m < 16; m <<= 1) v += __shfl_xor(v, m, 32);
  return v;
}

// ---------------------------------------------------------------------------
// Weight transpose + cast:  in [L, R, C] f32  ->  out [L, C, R] bf16
// ---------------------------------------------------------------------------
__global__ void tcast_kernel(const float* __restrict__ in, __bf16* __restrict__ out,
                             int R, int C, size_t total) {
  size_t idx = (size_t)blockIdx.x * blockDim.x + threadIdx.x;
  if (idx >= total) return;
  int c = (int)(idx % C);
  size_t t = idx / C;
  int r = (int)(t % R);
  int l = (int)(t / R);
  out[((size_t)l * C + c) * R + r] = (__bf16)in[idx];
}

// elementwise f32 -> bf16
__global__ void cast_kernel(const float* __restrict__ in, __bf16* __restrict__ out, size_t n) {
  size_t i = (size_t)blockIdx.x * blockDim.x + threadIdx.x;
  if (i < n) out[i] = (__bf16)in[i];
}

// ---------------------------------------------------------------------------
// WMMA GEMM:  C[M,N] = A[M,K](bf16) @ W[K,N] + bias,  W stored transposed
// as WT[N,K] bf16.  Each wave computes a 32x64 tile: 2 M-subtiles x 4
// N-subtiles = 8 WMMAs per 32-K step.  The K loop is software-pipelined
// with two named fragment sets so next-step loads are in flight while the
// current step's WMMAs execute (K is always a multiple of 64 -> no tail).
// Block = 4 waves -> 128x64 tile per block.
// MODE 0: f32 out   MODE 1: bf16 out   MODE 2: bf16 out + ReLU
// ---------------------------------------------------------------------------
template <int MODE>
__global__ void __launch_bounds__(128) gemm_kernel(
    const __bf16* __restrict__ A, const __bf16* __restrict__ WT,
    const float* __restrict__ bias, void* __restrict__ Cout, int N, int K) {
  const int wave = threadIdx.x >> 5;
  const int lane = threadIdx.x & 31;
  const int hlf  = lane >> 4;
  const int nl   = lane & 15;
  const int tn   = blockIdx.x * 64;
  const int tm   = (blockIdx.y * 4 + wave) * 32;

  const __bf16* Ar0 = A + (size_t)(tm + nl) * K;
  const __bf16* Ar1 = A + (size_t)(tm + 16 + nl) * K;
  const __bf16* B0 = WT + (size_t)(tn + nl) * K;
  const __bf16* B1 = WT + (size_t)(tn + 16 + nl) * K;
  const __bf16* B2 = WT + (size_t)(tn + 32 + nl) * K;
  const __bf16* B3 = WT + (size_t)(tn + 48 + nl) * K;

  v8f c00 = {}, c01 = {}, c02 = {}, c03 = {};
  v8f c10 = {}, c11 = {}, c12 = {}, c13 = {};

  ABfrag aA0, aA1, aB0, aB1;
  v16bf bA0, bA1, bA2, bA3, bB0, bB1, bB2, bB3;

  auto LD = [&](int k0, ABfrag& x0, ABfrag& x1, v16bf& y0, v16bf& y1,
                v16bf& y2, v16bf& y3) {
    x0.h[0] = *(const v8bf*)(Ar0 + k0 + hlf * 16);
    x0.h[1] = *(const v8bf*)(Ar0 + k0 + hlf * 16 + 8);
    x1.h[0] = *(const v8bf*)(Ar1 + k0 + hlf * 16);
    x1.h[1] = *(const v8bf*)(Ar1 + k0 + hlf * 16 + 8);
    y0 = *(const v16bf*)(B0 + k0 + hlf * 16);
    y1 = *(const v16bf*)(B1 + k0 + hlf * 16);
    y2 = *(const v16bf*)(B2 + k0 + hlf * 16);
    y3 = *(const v16bf*)(B3 + k0 + hlf * 16);
  };
  auto MMA = [&](ABfrag& x0, ABfrag& x1, v16bf& y0, v16bf& y1, v16bf& y2,
                 v16bf& y3) {
    c00 = __builtin_amdgcn_wmma_f32_16x16x32_bf16(false, x0.v, false, y0, (short)0, c00, false, false);
    c01 = __builtin_amdgcn_wmma_f32_16x16x32_bf16(false, x0.v, false, y1, (short)0, c01, false, false);
    c02 = __builtin_amdgcn_wmma_f32_16x16x32_bf16(false, x0.v, false, y2, (short)0, c02, false, false);
    c03 = __builtin_amdgcn_wmma_f32_16x16x32_bf16(false, x0.v, false, y3, (short)0, c03, false, false);
    c10 = __builtin_amdgcn_wmma_f32_16x16x32_bf16(false, x1.v, false, y0, (short)0, c10, false, false);
    c11 = __builtin_amdgcn_wmma_f32_16x16x32_bf16(false, x1.v, false, y1, (short)0, c11, false, false);
    c12 = __builtin_amdgcn_wmma_f32_16x16x32_bf16(false, x1.v, false, y2, (short)0, c12, false, false);
    c13 = __builtin_amdgcn_wmma_f32_16x16x32_bf16(false, x1.v, false, y3, (short)0, c13, false, false);
  };

  LD(0, aA0, aA1, bA0, bA1, bA2, bA3);
  for (int k0 = 0; k0 < K; k0 += 64) {
    LD(k0 + 32, aB0, aB1, bB0, bB1, bB2, bB3);       // in flight over set-A MMAs
    MMA(aA0, aA1, bA0, bA1, bA2, bA3);
    if (k0 + 64 < K) LD(k0 + 64, aA0, aA1, bA0, bA1, bA2, bA3);
    MMA(aB0, aB1, bB0, bB1, bB2, bB3);
  }

  auto STORE = [&](v8f& acc, int mt, int t) {
    float bc = bias[tn + t * 16 + nl];
#pragma unroll
    for (int r = 0; r < 8; ++r) {
      float v = acc[r] + bc;
      if (MODE == 2) v = fmaxf(v, 0.0f);
      size_t off = (size_t)(tm + mt * 16 + r + hlf * 8) * N + tn + t * 16 + nl;
      if (MODE == 0) ((float*)Cout)[off] = v;
      else           ((__bf16*)Cout)[off] = (__bf16)v;
    }
  };
  STORE(c00, 0, 0); STORE(c01, 0, 1); STORE(c02, 0, 2); STORE(c03, 0, 3);
  STORE(c10, 1, 0); STORE(c11, 1, 1); STORE(c12, 1, 2); STORE(c13, 1, 3);
}

// ---------------------------------------------------------------------------
// Flash attention, one (b,h) per blockIdx.y, 64 query rows per block,
// one 16-row query tile per wave (4 waves).  Key blocks of 32.
// scores = Q(16x32) @ K^T via WMMA; PV via WMMA with P repacked through LDS
// and V staged transposed in LDS.  Causal mask: key j allowed iff
// j <= i + mask_k - 1; fully-masked rows output zero (softmax * mask).
// ---------------------------------------------------------------------------
__global__ void __launch_bounds__(128) attn_kernel(
    const __bf16* __restrict__ Qg, const __bf16* __restrict__ Kg,
    const __bf16* __restrict__ Vg, __bf16* __restrict__ Og, int mask_k) {
  __shared__ __attribute__((aligned(64))) __bf16 Pb[4][16][32];
  __shared__ __attribute__((aligned(64))) __bf16 VT[4][32][48];  // [dk][key], 96B rows

  const int wave = threadIdx.x >> 5;
  const int lane = threadIdx.x & 31;
  const int hlf  = lane >> 4;
  const int nl   = lane & 15;
  const int bh   = blockIdx.y;
  const int b    = bh >> 3;
  const int h    = bh & 7;
  const int qt   = blockIdx.x * 64 + wave * 16;
  const size_t base = ((size_t)b * SS) * DDIM + (size_t)h * DKH;

  ABfrag qa;
  {
    const __bf16* qrow = Qg + base + (size_t)(qt + nl) * DDIM;
    qa.h[0] = *(const v8bf*)(qrow + hlf * 16);
    qa.h[1] = *(const v8bf*)(qrow + hlf * 16 + 8);
  }

  v8f acc0 = {}, acc1 = {};
  float rmax[8], rsum[8];
#pragma unroll
  for (int r = 0; r < 8; ++r) { rmax[r] = -3.0e38f; rsum[r] = 0.0f; }

  const float scale = 0.17677669529663687f;  // 1/sqrt(32)
  const int jmax = qt + 15 + mask_k - 1;     // >= 14 always

  for (int j0 = 0; j0 <= jmax; j0 += 32) {
    // stage V^T[dk][key] for this key block
    {
      const __bf16* vrow = Vg + base + (size_t)(j0 + lane) * DDIM;
#pragma unroll
      for (int d = 0; d < DKH; ++d) VT[wave][d][lane] = vrow[d];
    }
    // K fragments (B layout: N=key column, contiguous K run per lane)
    v16bf kb0, kb1;
    {
      const __bf16* k0r = Kg + base + (size_t)(j0 + nl) * DDIM;
      const __bf16* k1r = Kg + base + (size_t)(j0 + 16 + nl) * DDIM;
      kb0 = *(const v16bf*)(k0r + hlf * 16);
      kb1 = *(const v16bf*)(k1r + hlf * 16);
    }
    v8f s0 = {}, s1 = {};
    s0 = __builtin_amdgcn_wmma_f32_16x16x32_bf16(false, qa.v, false, kb0, (short)0, s0, false, false);
    s1 = __builtin_amdgcn_wmma_f32_16x16x32_bf16(false, qa.v, false, kb1, (short)0, s1, false, false);

#pragma unroll
    for (int r = 0; r < 8; ++r) {
      const int qi  = qt + r + hlf * 8;
      const int lim = qi + mask_k - 1;
      float a0 = (j0 + nl <= lim)      ? s0[r] * scale : -1.0e32f;
      float a1 = (j0 + 16 + nl <= lim) ? s1[r] * scale : -1.0e32f;
      float bmax = redmax16(fmaxf(a0, a1));
      float nm = fmaxf(rmax[r], bmax);
      bool valid = nm > -1.0e30f;
      float corr = valid ? __expf(rmax[r] - nm) : 1.0f;
      float p0 = valid ? __expf(a0 - nm) : 0.0f;
      float p1 = valid ? __expf(a1 - nm) : 0.0f;
      rsum[r] = rsum[r] * corr + redsum16(p0 + p1);
      rmax[r] = nm;
      acc0[r] *= corr;
      acc1[r] *= corr;
      Pb[wave][r + hlf * 8][nl]      = (__bf16)p0;
      Pb[wave][r + hlf * 8][16 + nl] = (__bf16)p1;
    }
    __asm__ volatile("" ::: "memory");  // keep LDS W->R order (in-order per wave)

    ABfrag pa;
    pa.h[0] = *(const v8bf*)&Pb[wave][nl][hlf * 16];
    pa.h[1] = *(const v8bf*)&Pb[wave][nl][hlf * 16 + 8];
    v16bf vb0 = *(const v16bf*)&VT[wave][nl][hlf * 16];
    v16bf vb1 = *(const v16bf*)&VT[wave][16 + nl][hlf * 16];
    acc0 = __builtin_amdgcn_wmma_f32_16x16x32_bf16(false, pa.v, false, vb0, (short)0, acc0, false, false);
    acc1 = __builtin_amdgcn_wmma_f32_16x16x32_bf16(false, pa.v, false, vb1, (short)0, acc1, false, false);
  }

#pragma unroll
  for (int r = 0; r < 8; ++r) {
    float inv = rsum[r] > 0.0f ? 1.0f / rsum[r] : 0.0f;
    size_t o = base + (size_t)(qt + r + hlf * 8) * DDIM;
    Og[o + nl]      = (__bf16)(acc0[r] * inv);
    Og[o + 16 + nl] = (__bf16)(acc1[r] * inv);
  }
}

// ---------------------------------------------------------------------------
// out = LayerNorm(res + y) * g + b  -> f32 state + bf16 copy
// one wave per row of D=256 (8 elems/lane), 8 rows per 256-thread block
// ---------------------------------------------------------------------------
__global__ void __launch_bounds__(256) add_ln_kernel(
    const float* __restrict__ res, const float* __restrict__ y,
    const float* __restrict__ g, const float* __restrict__ beta,
    float* __restrict__ outf, __bf16* __restrict__ outb) {
  const int wave = threadIdx.x >> 5;
  const int lane = threadIdx.x & 31;
  const size_t row = (size_t)blockIdx.x * 8 + wave;
  const float* xr = res + row * DDIM;
  const float* yr = y + row * DDIM;

  float v[8];
  float s = 0.0f;
#pragma unroll
  for (int i = 0; i < 8; ++i) {
    v[i] = xr[lane + i * 32] + yr[lane + i * 32];
    s += v[i];
  }
#pragma unroll
  for (int m = 1; m < 32; m <<= 1) s += __shfl_xor(s, m, 32);
  const float mean = s * (1.0f / DDIM);
  float q = 0.0f;
#pragma unroll
  for (int i = 0; i < 8; ++i) { float d = v[i] - mean; q += d * d; }
#pragma unroll
  for (int m = 1; m < 32; m <<= 1) q += __shfl_xor(q, m, 32);
  const float rstd = rsqrtf(q * (1.0f / DDIM) + 1e-5f);

#pragma unroll
  for (int i = 0; i < 8; ++i) {
    int c = lane + i * 32;
    float o = (v[i] - mean) * rstd * g[c] + beta[c];
    outf[row * DDIM + c] = o;
    outb[row * DDIM + c] = (__bf16)o;
  }
}

// ---------------------------------------------------------------------------
extern "C" void kernel_launch(void* const* d_in, const int* in_sizes, int n_in,
                              void* d_out, int out_size, void* d_ws, size_t ws_size,
                              hipStream_t stream) {
  const float* q_embed  = (const float*)d_in[0];
  const float* qa_embed = (const float*)d_in[1];
  const float* Wk  = (const float*)d_in[2];
  const float* bk  = (const float*)d_in[3];
  const float* Wv  = (const float*)d_in[4];
  const float* bv  = (const float*)d_in[5];
  const float* Wo  = (const float*)d_in[6];
  const float* bo  = (const float*)d_in[7];
  const float* l1g = (const float*)d_in[8];
  const float* l1b = (const float*)d_in[9];
  const float* W1  = (const float*)d_in[10];
  const float* b1  = (const float*)d_in[11];
  const float* W2  = (const float*)d_in[12];
  const float* b2  = (const float*)d_in[13];
  const float* l2g = (const float*)d_in[14];
  const float* l2b = (const float*)d_in[15];

  // ---- workspace carve-out ----
  char* p = (char*)d_ws;
  auto carve = [&](size_t bytes) -> void* {
    void* r = (void*)p;
    p += (bytes + 255) & ~(size_t)255;
    return r;
  };
  __bf16* WkT = (__bf16*)carve((size_t)LLAY * DDIM * DDIM * 2);
  __bf16* WvT = (__bf16*)carve((size_t)LLAY * DDIM * DDIM * 2);
  __bf16* WoT = (__bf16*)carve((size_t)LLAY * DDIM * DDIM * 2);
  __bf16* W1T = (__bf16*)carve((size_t)LLAY * DDIM * DFFN * 2);
  __bf16* W2T = (__bf16*)carve((size_t)LLAY * DFFN * DDIM * 2);
  float*  ys  = (float*)carve((size_t)NROWS * DDIM * 4);
  float*  xs  = (float*)carve((size_t)NROWS * DDIM * 4);
  float*  tf  = (float*)carve((size_t)NROWS * DDIM * 4);
  float*  prj = (float*)carve((size_t)NROWS * DDIM * 4);
  __bf16* yb  = (__bf16*)carve((size_t)NROWS * DDIM * 2);
  __bf16* xb  = (__bf16*)carve((size_t)NROWS * DDIM * 2);
  __bf16* tb  = (__bf16*)carve((size_t)NROWS * DDIM * 2);
  __bf16* qb  = (__bf16*)carve((size_t)NROWS * DDIM * 2);
  __bf16* kb  = (__bf16*)carve((size_t)NROWS * DDIM * 2);
  __bf16* vb  = (__bf16*)carve((size_t)NROWS * DDIM * 2);
  __bf16* ab  = (__bf16*)carve((size_t)NROWS * DDIM * 2);
  __bf16* hb  = (__bf16*)carve((size_t)NROWS * DFFN * 2);

  // ---- weights: transpose + cast to bf16 [N,K] ----
  {
    size_t t = (size_t)LLAY * DDIM * DDIM;
    int blks = (int)((t + 255) / 256);
    tcast_kernel<<<blks, 256, 0, stream>>>(Wk, WkT, DDIM, DDIM, t);
    tcast_kernel<<<blks, 256, 0, stream>>>(Wv, WvT, DDIM, DDIM, t);
    tcast_kernel<<<blks, 256, 0, stream>>>(Wo, WoT, DDIM, DDIM, t);
    size_t t1 = (size_t)LLAY * DDIM * DFFN;
    int blks1 = (int)((t1 + 255) / 256);
    tcast_kernel<<<blks1, 256, 0, stream>>>(W1, W1T, DDIM, DFFN, t1);
    tcast_kernel<<<blks1, 256, 0, stream>>>(W2, W2T, DFFN, DDIM, t1);
  }
  // ---- initial activation casts ----
  {
    size_t n = (size_t)NROWS * DDIM;
    int blks = (int)((n + 255) / 256);
    cast_kernel<<<blks, 256, 0, stream>>>(qa_embed, yb, n);
    cast_kernel<<<blks, 256, 0, stream>>>(q_embed, xb, n);
  }

  const dim3 gD(DDIM / 64, NROWS / 128), gF(DFFN / 64, NROWS / 128);
  const dim3 gA(SS / 64, BB * HH);

  auto layer = [&](int li, const float* qf, const __bf16* qsrc, const __bf16* ksrc,
                   const __bf16* vsrc, float* outf, __bf16* outb, int mask_k, bool pos) {
    const __bf16* wk = WkT + (size_t)li * DDIM * DDIM;
    const __bf16* wv = WvT + (size_t)li * DDIM * DDIM;
    const __bf16* wo = WoT + (size_t)li * DDIM * DDIM;
    gemm_kernel<1><<<gD, 128, 0, stream>>>(qsrc, wk, bk + li * DDIM, qb, DDIM, DDIM);
    gemm_kernel<1><<<gD, 128, 0, stream>>>(ksrc, wk, bk + li * DDIM, kb, DDIM, DDIM);
    gemm_kernel<1><<<gD, 128, 0, stream>>>(vsrc, wv, bv + li * DDIM, vb, DDIM, DDIM);
    attn_kernel<<<gA, 128, 0, stream>>>(qb, kb, vb, ab, mask_k);
    gemm_kernel<0><<<gD, 128, 0, stream>>>(ab, wo, bo + li * DDIM, prj, DDIM, DDIM);
    add_ln_kernel<<<NROWS / 8, 256, 0, stream>>>(qf, prj, l1g + li * DDIM, l1b + li * DDIM,
                                                 pos ? tf : outf, pos ? tb : outb);
    if (pos) {
      gemm_kernel<2><<<gF, 128, 0, stream>>>(tb, W1T + (size_t)li * DDIM * DFFN,
                                             b1 + li * DFFN, hb, DFFN, DDIM);
      gemm_kernel<0><<<gD, 128, 0, stream>>>(hb, W2T + (size_t)li * DFFN * DDIM,
                                             b2 + li * DDIM, prj, DDIM, DFFN);
      add_ln_kernel<<<NROWS / 8, 256, 0, stream>>>(tf, prj, l2g + li * DDIM, l2b + li * DDIM,
                                                   outf, outb);
    }
  };

  // encoders (self-attn, causal mask_k=1, with FFN)
  layer(0, qa_embed, yb, yb, yb, ys, yb, 1, true);
  layer(1, ys, yb, yb, yb, ys, yb, 1, true);
  // decoders: self (mask_k=1, no FFN) then cross (Q,K from x; V from y; mask_k=0, FFN)
  layer(2, q_embed, xb, xb, xb, xs, xb, 1, false);
  layer(3, xs, xb, xb, yb, xs, xb, 0, true);
  layer(4, xs, xb, xb, xb, xs, xb, 1, false);
  layer(5, xs, xb, xb, yb, (float*)d_out, xb, 0, true);  // final LN writes d_out
}
